// Final_model_49624052138771
// MI455X (gfx1250) — compile-verified, matrix operand order checked by reading, not run on hardware
//
#include <hip/hip_runtime.h>
#include <hip/hip_bf16.h>

// ---------------------------------------------------------------------------
// CDNA5 (gfx1250) bf16-WMMA implementation of the char-CNN/LSTM LM forward.
// All matmuls route through v_wmma_f32_16x16x32_bf16 (wave32, f32 accum).
// GEMM uses double-buffered LDS panels; global->LDS staging goes through
// GLOBAL_LOAD_ASYNC_TO_LDS_B128 (ASYNCcnt) when the builtin is available.
// ---------------------------------------------------------------------------

typedef __attribute__((ext_vector_type(16))) __bf16 v16bf;
typedef __attribute__((ext_vector_type(8)))  float  v8f;
typedef __attribute__((ext_vector_type(4)))  int    v4i;

// address-space-qualified pointer typedefs (attribute-in-cast is unreliable)
typedef __attribute__((address_space(1))) v4i* gas_v4i_p;
typedef __attribute__((address_space(3))) v4i* las_v4i_p;

#define NWORD 4096      // NUM_BLK * BLK_LEN
#define NUM_BLK 32
#define BLK_LEN 128
#define IND 1920
#define HID 256
#define VOCAB 32000
#define OUTF_ELEMS ((size_t)NWORD * VOCAB)   // 131072000 floats

#if defined(__gfx1250__) && __has_builtin(__builtin_amdgcn_global_load_async_to_lds_b128) && \
    __has_builtin(__builtin_amdgcn_s_wait_asynccnt)
#define ASYNC_LDS 1
#endif

__device__ __forceinline__ float sigf(float x) { return 1.0f / (1.0f + __expf(-x)); }

__device__ __forceinline__ v8f wmma_bf(v16bf a, v16bf b, v8f c) {
    // (neg_a, A, neg_b, B, c_mod, C, reuse_a, reuse_b)
    return __builtin_amdgcn_wmma_f32_16x16x32_bf16(false, a, false, b, (short)0, c, false, false);
}

// K index inside a 32-wide K panel for 16-bit A/B fragments (per ISA 7.12.2):
// lanes 0-15 hold K {0..7,16..23}, lanes 16-31 hold K {8..15,24..31}.
__device__ __forceinline__ int kmap(int lane, int h) {
    return h + (h & 8) + ((lane >> 1) & 8);
}

// 16-byte global -> LDS copy (async when available)
__device__ __forceinline__ void cp16(const __bf16* g, __bf16* l) {
#ifdef ASYNC_LDS
    __builtin_amdgcn_global_load_async_to_lds_b128((gas_v4i_p)g, (las_v4i_p)l, 0, 0);
#else
    *(int4*)l = *(const int4*)g;
#endif
}

template <int N>
__device__ __forceinline__ void async_wait() {
#ifdef ASYNC_LDS
    __builtin_amdgcn_s_wait_asynccnt(N);
#endif
}

// --------------------------- f32 -> bf16 cast ------------------------------
__global__ void k_cast_bf16(const float* __restrict__ s, __bf16* __restrict__ d, int n) {
    int i = blockIdx.x * 256 + threadIdx.x;
    if (i < n) d[i] = (__bf16)s[i];
}

// --------------------------- char embedding --------------------------------
// out: [word][char(16)][emb(16)] bf16, word-major contiguous 256 per word
__global__ void k_embed(const int* __restrict__ ids, const float* __restrict__ tbl,
                        __bf16* __restrict__ out) {
    int i = blockIdx.x * 256 + threadIdx.x;           // 4096*256
    int w = i >> 8, r = (i >> 4) & 15, e = i & 15;
    int id = ids[w * 16 + r];
    out[i] = (__bf16)tbl[id * 16 + e];
}

// --------------------------- conv (WMMA) -----------------------------------
// For width FW: y[w][p][n] = sum_{k<16*FW} emb[w][16p+k] * W[k][n]; feature =
// max_p tanh(y + b). One block: 16 words x 128 channels, 8 waves (1 tile each).
template <int FW>
__global__ __launch_bounds__(256) void k_conv(const __bf16* __restrict__ emb,
                                              const __bf16* __restrict__ Wc,
                                              const float* __restrict__ bias,
                                              float* __restrict__ ccf,
                                              __bf16* __restrict__ ccb,
                                              int fn, int conv_off) {
    constexpr int KW = FW * 16;
    constexpr int P  = 17 - FW;
    constexpr int KT = (KW + 31) / 32;
    __shared__ __attribute__((aligned(16))) __bf16 lE[16][288];      // zero-padded tail
    __shared__ __attribute__((aligned(16))) __bf16 lW[KT * 32][128]; // zero-padded K

    int wt = blockIdx.y;                  // word tile (16 words)
    int n0 = blockIdx.x * 128;            // channel base
    int tid = threadIdx.x;

    for (int e = tid; e < 16 * 288; e += 256) {
        int w = e / 288, c = e % 288;
        lE[w][c] = (c < 256) ? emb[(size_t)(wt * 16 + w) * 256 + c] : (__bf16)0.0f;
    }
    for (int e = tid; e < KT * 32 * 128; e += 256) {
        int k = e >> 7, n = e & 127;
        lW[k][n] = (k < KW) ? Wc[(size_t)k * fn + n0 + n] : (__bf16)0.0f;
    }
    __syncthreads();

    int wave = tid >> 5, lane = tid & 31;
    int lr = lane & 15;
    int hi = (lane & 16) ? 8 : 0;
    int co = wave * 16;

    v16bf bfrag[KT];
#pragma unroll
    for (int kt = 0; kt < KT; ++kt)
#pragma unroll
        for (int h = 0; h < 16; ++h)
            bfrag[kt][h] = lW[kt * 32 + kmap(lane, h)][co + lr];

    float mx[8];
#pragma unroll
    for (int r = 0; r < 8; ++r) mx[r] = -1e30f;
    float bv = bias[n0 + co + lr];

    for (int p = 0; p < P; ++p) {
        v8f acc = {};
#pragma unroll
        for (int kt = 0; kt < KT; ++kt) {
            v16bf a;
#pragma unroll
            for (int h = 0; h < 16; ++h)
                a[h] = lE[lr][16 * p + kt * 32 + kmap(lane, h)];
            acc = wmma_bf(a, bfrag[kt], acc);
        }
#pragma unroll
        for (int r = 0; r < 8; ++r)
            mx[r] = fmaxf(mx[r], tanhf(acc[r] + bv));
    }
#pragma unroll
    for (int r = 0; r < 8; ++r) {
        int word = wt * 16 + r + hi;
        int n = conv_off + n0 + co + lr;
        size_t idx = (size_t)word * IND + n;
        ccf[idx] = mx[r];
        ccb[idx] = (__bf16)mx[r];
    }
}

// --------------------------- generic GEMM (WMMA) ---------------------------
// C[m,n] = act( sum_k A[m,k] * W[n,k] + bias[n] + Cinit[m,n] )
// A: bf16 [M,K] row-major; W: bf16 [N,K] row-major (torch Linear weight).
// Block tile 32 x (64*WN): 8 waves as 2(M) x 4(N), WN 16-col tiles per wave.
// Double-buffered LDS K-panels (32 wide), async global->LDS staging.
// Requires M%32==0, N%(64*WN)==0, K%32==0. act: 0 none,1 relu,2 sigmoid,3 tanh.
template <int WN>
__global__ __launch_bounds__(256) void k_gemm_bt(const __bf16* __restrict__ A,
                                                 const __bf16* __restrict__ W,
                                                 const float* __restrict__ bias,
                                                 const float* __restrict__ Cinit,
                                                 float* __restrict__ Cf,
                                                 __bf16* __restrict__ Cbf,
                                                 int M, int N, int K, int act) {
    constexpr int BN = 64 * WN;
    __shared__ __attribute__((aligned(16))) __bf16 lA[2][32][32];
    __shared__ __attribute__((aligned(16))) __bf16 lW[2][BN][32];
    int gm0 = blockIdx.y * 32;
    int gn0 = blockIdx.x * BN;
    int tid = threadIdx.x;
    int wave = tid >> 5, lane = tid & 31;
    int ro = (wave & 1) * 16;                 // wave row base (0/16)
    int cw = (wave >> 1) * (16 * WN);         // wave col base
    int lr = lane & 15;
    int hi = (lane & 16) ? 8 : 0;

    v8f acc[WN];
#pragma unroll
    for (int j = 0; j < WN; ++j) acc[j] = (v8f){};
    if (Cinit) {
#pragma unroll
        for (int j = 0; j < WN; ++j)
#pragma unroll
            for (int r = 0; r < 8; ++r)
                acc[j][r] = Cinit[(size_t)(gm0 + ro + r + hi) * N + gn0 + cw + j * 16 + lr];
    }

    int KT = K >> 5;
    // one K-panel stage: WN W-chunks + 1 A-chunk per thread (16B each)
    auto stage = [&](int kt, int buf) {
        int k0 = kt << 5;
#pragma unroll
        for (int i = 0; i < WN; ++i) {
            int c = tid * WN + i;             // 0 .. 256*WN-1 == BN*4-1
            int r = c >> 2, kq = (c & 3) * 8;
            cp16(&W[(size_t)(gn0 + r) * K + k0 + kq], &lW[buf][r][kq]);
        }
        int c = tid & 127;                    // 32 rows * 4 chunks (dup tid>=128)
        int r = c >> 2, kq = (c & 3) * 8;
        cp16(&A[(size_t)(gm0 + r) * K + k0 + kq], &lA[buf][r][kq]);
    };

    stage(0, 0);
    for (int kt = 0; kt < KT; ++kt) {
        int cur = kt & 1;
        if (kt + 1 < KT) {
            stage(kt + 1, cur ^ 1);
            async_wait<WN + 1>();             // previous panel's batch complete
        } else {
            async_wait<0>();
        }
        __syncthreads();                      // panel `cur` visible to all waves

        v16bf a, b;
#pragma unroll
        for (int h = 0; h < 16; ++h) a[h] = lA[cur][ro + lr][kmap(lane, h)];
#pragma unroll
        for (int j = 0; j < WN; ++j) {
#pragma unroll
            for (int h = 0; h < 16; ++h) b[h] = lW[cur][cw + j * 16 + lr][kmap(lane, h)];
            acc[j] = wmma_bf(a, b, acc[j]);
        }
        __syncthreads();                      // done reading before next overwrite
    }

#pragma unroll
    for (int j = 0; j < WN; ++j) {
#pragma unroll
        for (int r = 0; r < 8; ++r) {
            int m = gm0 + ro + r + hi;
            int n = gn0 + cw + j * 16 + lr;
            float v = acc[j][r];
            if (bias) v += bias[n];
            if (act == 1)      v = fmaxf(v, 0.0f);
            else if (act == 2) v = sigf(v);
            else if (act == 3) v = tanhf(v);
            size_t idx = (size_t)m * N + n;
            if (Cf)  Cf[idx]  = v;
            if (Cbf) Cbf[idx] = (__bf16)v;
        }
    }
}

// --------------------------- highway mix -----------------------------------
// hw = gate*nonl + (1-gate)*cc ; written bf16 transposed to [t][b][1920]
__global__ void k_hw_mix(const float* __restrict__ nonl, const float* __restrict__ gate,
                         const float* __restrict__ cc, __bf16* __restrict__ hwT) {
    int i = blockIdx.x * 256 + threadIdx.x;      // 4096*1920
    int m = i / IND, c = i % IND;
    int b = m >> 7, t = m & 127;
    float g = gate[i];
    float v = g * nonl[i] + (1.0f - g) * cc[i];
    hwT[((size_t)t * NUM_BLK + b) * IND + c] = (__bf16)v;
}

__global__ void k_bias_sum(const float* a, const float* b, float* o, int n) {
    int i = blockIdx.x * 256 + threadIdx.x;
    if (i < n) o[i] = a[i] + b[i];
}

// --------------------------- LSTM cell -------------------------------------
__global__ void k_lstm_cell(const float* __restrict__ gates, float* __restrict__ cbuf,
                            __bf16* __restrict__ hbf, float* __restrict__ outblks, int t) {
    int i = blockIdx.x * 256 + threadIdx.x;      // 32*256
    int b = i >> 8, j = i & 255;
    const float* g = gates + (size_t)b * 1024;
    float ig = g[j], fg = g[256 + j], gg = g[512 + j], og = g[768 + j];
    float c = sigf(fg) * cbuf[i] + sigf(ig) * tanhf(gg);
    float h = sigf(og) * tanhf(c);
    cbuf[i] = c;
    hbf[i] = (__bf16)h;
    outblks[((size_t)b * BLK_LEN + t) * HID + j] = h;
}

// --------------------------- pooling ---------------------------------------
__global__ void k_pool(const float* __restrict__ outblks, __bf16* __restrict__ pooled) {
    int i = blockIdx.x * 256 + threadIdx.x;      // 32*128*128
    int bt = i >> 7, j = i & 127;
    float a = outblks[(size_t)bt * HID + 2 * j];
    float c = outblks[(size_t)bt * HID + 2 * j + 1];
    pooled[(size_t)bt * HID + j]       = (__bf16)fmaxf(a, c);
    pooled[(size_t)bt * HID + 128 + j] = (__bf16)(0.5f * (a + c));
}

// --------------------------- global LSTM (tiny, VALU) ----------------------
__global__ __launch_bounds__(256) void k_glstm(const float* __restrict__ x,
                                               const float* __restrict__ Wih,
                                               const float* __restrict__ Whh,
                                               const float* __restrict__ bih,
                                               const float* __restrict__ bhh,
                                               float* __restrict__ outg) {
    __shared__ float hs[256];
    int j = threadIdx.x;
    hs[j] = 0.0f;
    float c = 0.0f;
    float bsum[4];
#pragma unroll
    for (int g = 0; g < 4; ++g) bsum[g] = bih[g * 256 + j] + bhh[g * 256 + j];
    __syncthreads();
    for (int t = 0; t < 32; ++t) {
        const float* xt = x + (size_t)t * 256;
        float acc[4];
#pragma unroll
        for (int g = 0; g < 4; ++g) {
            const float* wi = Wih + (size_t)(g * 256 + j) * 256;
            const float* wh = Whh + (size_t)(g * 256 + j) * 256;
            float s = bsum[g];
            for (int k = 0; k < 256; ++k) s += xt[k] * wi[k] + hs[k] * wh[k];
            acc[g] = s;
        }
        __syncthreads();
        float cn = sigf(acc[1]) * c + sigf(acc[0]) * tanhf(acc[2]);
        float hn = sigf(acc[3]) * tanhf(cn);
        c = cn;
        hs[j] = hn;
        outg[(size_t)t * 256 + j] = hn;
        __syncthreads();
    }
}

// --------------------------- concat for fc ---------------------------------
__global__ void k_concat(const float* __restrict__ outblks, const float* __restrict__ outg,
                         __bf16* __restrict__ x) {
    int i = blockIdx.x * 256 + threadIdx.x;      // 3968*512
    int m = i >> 9, c = i & 511;
    int bb = (m >> 7) + 1, t = m & 127;
    float v = (c < 256) ? outblks[((size_t)bb * BLK_LEN + t) * HID + c]
                        : outg[(size_t)(bb - 1) * 256 + (c - 256)];
    x[i] = (__bf16)v;
}

// ===========================================================================
extern "C" void kernel_launch(void* const* d_in, const int* in_sizes, int n_in,
                              void* d_out, int out_size, void* d_ws, size_t ws_size,
                              hipStream_t stream) {
    const int*   inputs     = (const int*)d_in[0];
    const float* char_embed = (const float*)d_in[1];
    const float* conv_w[4]  = { (const float*)d_in[2], (const float*)d_in[4],
                                (const float*)d_in[6], (const float*)d_in[8] };
    const float* conv_b[4]  = { (const float*)d_in[3], (const float*)d_in[5],
                                (const float*)d_in[7], (const float*)d_in[9] };
    const float* hw_nonl_w = (const float*)d_in[10];
    const float* hw_nonl_b = (const float*)d_in[11];
    const float* hw_gate_w = (const float*)d_in[12];
    const float* hw_gate_b = (const float*)d_in[13];
    const float* lstm_Wih  = (const float*)d_in[14];
    const float* lstm_Whh  = (const float*)d_in[15];
    const float* lstm_bih  = (const float*)d_in[16];
    const float* lstm_bhh  = (const float*)d_in[17];
    const float* fc1_w     = (const float*)d_in[18];
    const float* fc1_b     = (const float*)d_in[19];
    const float* g_Wih     = (const float*)d_in[20];
    const float* g_Whh     = (const float*)d_in[21];
    const float* g_bih     = (const float*)d_in[22];
    const float* g_bhh     = (const float*)d_in[23];
    const float* fc_w      = (const float*)d_in[24];
    const float* fc_b      = (const float*)d_in[25];
    const float* fc_out_w  = (const float*)d_in[26];
    const float* fc_out_b  = (const float*)d_in[27];

    float* outf    = (float*)d_out;                    // [32,128,32000]
    float* outblks = (float*)d_out + OUTF_ELEMS;       // [32,128,256]

    // ---- workspace carve-out ----
    char* ws = (char*)d_ws;
    size_t off = 0;
    auto alloc = [&](size_t bytes) -> void* {
        void* p = ws + off;
        off += (bytes + 255) & ~(size_t)255;
        return p;
    };
    const int fn[4] = { 128, 256, 512, 1024 };
    const int fw[4] = { 2, 3, 4, 5 };
    const int coff[4] = { 0, 128, 384, 896 };

    __bf16* cw_bf[4];
    for (int i = 0; i < 4; ++i) cw_bf[i] = (__bf16*)alloc((size_t)fw[i] * 16 * fn[i] * 2);
    __bf16* hwN_bf = (__bf16*)alloc((size_t)IND * IND * 2);
    __bf16* hwG_bf = (__bf16*)alloc((size_t)IND * IND * 2);
    __bf16* Wih_bf = (__bf16*)alloc((size_t)1024 * IND * 2);
    __bf16* Whh_bf = (__bf16*)alloc((size_t)1024 * HID * 2);
    __bf16* fc1w_bf = (__bf16*)alloc((size_t)256 * 32768 * 2);
    __bf16* fcw_bf  = (__bf16*)alloc((size_t)256 * 512 * 2);
    __bf16* fow_bf  = (__bf16*)alloc((size_t)VOCAB * HID * 2);

    __bf16* emb_bf = (__bf16*)alloc((size_t)NWORD * 256 * 2);
    float*  cc_f   = (float*)alloc((size_t)NWORD * IND * 4);
    __bf16* cc_bf  = (__bf16*)alloc((size_t)NWORD * IND * 2);
    float*  nonl_f = (float*)alloc((size_t)NWORD * IND * 4);
    float*  gate_f = (float*)alloc((size_t)NWORD * IND * 4);
    __bf16* hwT_bf = (__bf16*)alloc((size_t)NWORD * IND * 2);
    float*  XG     = (float*)alloc((size_t)NWORD * 1024 * 4);
    float*  gates  = (float*)alloc((size_t)32 * 1024 * 4);
    __bf16* h_bf   = (__bf16*)alloc((size_t)32 * 256 * 2);
    float*  c_f    = (float*)alloc((size_t)32 * 256 * 4);
    __bf16* pooled = (__bf16*)alloc((size_t)32 * 32768 * 2);
    float*  blkrep = (float*)alloc((size_t)32 * 256 * 4);
    float*  out_g  = (float*)alloc((size_t)32 * 256 * 4);
    __bf16* xcat   = (__bf16*)alloc((size_t)3968 * 512 * 2);
    __bf16* outf_bf = (__bf16*)alloc((size_t)NWORD * 256 * 2);
    float*  bsum   = (float*)alloc((size_t)1024 * 4);
    (void)ws_size; (void)n_in; (void)in_sizes; (void)out_size;

    auto cast = [&](const float* s, __bf16* d, int n) {
        k_cast_bf16<<<(n + 255) / 256, 256, 0, stream>>>(s, d, n);
    };
    for (int i = 0; i < 4; ++i) cast(conv_w[i], cw_bf[i], fw[i] * 16 * fn[i]);
    cast(hw_nonl_w, hwN_bf, IND * IND);
    cast(hw_gate_w, hwG_bf, IND * IND);
    cast(lstm_Wih, Wih_bf, 1024 * IND);
    cast(lstm_Whh, Whh_bf, 1024 * HID);
    cast(fc1_w, fc1w_bf, 256 * 32768);
    cast(fc_w, fcw_bf, 256 * 512);
    cast(fc_out_w, fow_bf, VOCAB * HID);

    // 1) embedding gather
    k_embed<<<NWORD, 256, 0, stream>>>(inputs, char_embed, emb_bf);

    // 2) convs -> cc [4096,1920] (f32 + bf16)
    k_conv<2><<<dim3(fn[0] / 128, 256), 256, 0, stream>>>(emb_bf, cw_bf[0], conv_b[0], cc_f, cc_bf, fn[0], coff[0]);
    k_conv<3><<<dim3(fn[1] / 128, 256), 256, 0, stream>>>(emb_bf, cw_bf[1], conv_b[1], cc_f, cc_bf, fn[1], coff[1]);
    k_conv<4><<<dim3(fn[2] / 128, 256), 256, 0, stream>>>(emb_bf, cw_bf[2], conv_b[2], cc_f, cc_bf, fn[2], coff[2]);
    k_conv<5><<<dim3(fn[3] / 128, 256), 256, 0, stream>>>(emb_bf, cw_bf[3], conv_b[3], cc_f, cc_bf, fn[3], coff[3]);

    // 3) highway: nonl = relu(cc@Wn^T+b), gate = sigmoid(cc@Wg^T+b), mix
    k_gemm_bt<2><<<dim3(IND / 128, NWORD / 32), 256, 0, stream>>>(cc_bf, hwN_bf, hw_nonl_b, nullptr,
                                                                  nonl_f, nullptr, NWORD, IND, IND, 1);
    k_gemm_bt<2><<<dim3(IND / 128, NWORD / 32), 256, 0, stream>>>(cc_bf, hwG_bf, hw_gate_b, nullptr,
                                                                  gate_f, nullptr, NWORD, IND, IND, 2);
    k_hw_mix<<<(NWORD * IND) / 256, 256, 0, stream>>>(nonl_f, gate_f, cc_f, hwT_bf);

    // 4) local LSTM: precompute x-gates XG[t*32+b, 1024] = hwT@Wih^T + (bih+bhh)
    k_bias_sum<<<4, 256, 0, stream>>>(lstm_bih, lstm_bhh, bsum, 1024);
    k_gemm_bt<4><<<dim3(1024 / 256, NWORD / 32), 256, 0, stream>>>(hwT_bf, Wih_bf, bsum, nullptr,
                                                                   XG, nullptr, NWORD, 1024, IND, 0);
    (void)hipMemsetAsync(h_bf, 0, (size_t)32 * 256 * 2, stream);
    (void)hipMemsetAsync(c_f, 0, (size_t)32 * 256 * 4, stream);
    for (int t = 0; t < BLK_LEN; ++t) {
        k_gemm_bt<4><<<dim3(1024 / 256, 1), 256, 0, stream>>>(h_bf, Whh_bf, nullptr,
                                                              XG + (size_t)t * 32 * 1024,
                                                              gates, nullptr, 32, 1024, 256, 0);
        k_lstm_cell<<<32, 256, 0, stream>>>(gates, c_f, h_bf, outblks, t);
    }

    // 5) pooling -> pooled bf16 [32, 32768]
    k_pool<<<(32 * 128 * 128) / 256, 256, 0, stream>>>(outblks, pooled);

    // 6) fc1 -> blkrep [32,256]
    k_gemm_bt<4><<<dim3(1, 1), 256, 0, stream>>>(pooled, fc1w_bf, fc1_b, nullptr,
                                                 blkrep, nullptr, 32, 256, 32768, 0);

    // 7) global LSTM (tiny) -> out_g [32,256]
    k_glstm<<<1, 256, 0, stream>>>(blkrep, g_Wih, g_Whh, g_bih, g_bhh, out_g);

    // 8) fc over concat(outblks[1:], glob) -> outf_bf rows 128..4095; rows 0..127 zero
    k_concat<<<(3968 * 512) / 256, 256, 0, stream>>>(outblks, out_g, xcat);
    (void)hipMemsetAsync(outf_bf, 0, (size_t)128 * 256 * 2, stream);
    k_gemm_bt<4><<<dim3(1, 3968 / 32), 256, 0, stream>>>(xcat, fcw_bf, fc_b, nullptr,
                                                         nullptr, outf_bf + (size_t)128 * 256,
                                                         3968, 256, 512, 0);

    // 9) fc_out -> d_out [4096, 32000] f32 (zero rows -> bias, matching reference)
    k_gemm_bt<4><<<dim3(VOCAB / 256, NWORD / 32), 256, 0, stream>>>(outf_bf, fow_bf, fc_out_b, nullptr,
                                                                    outf, nullptr, NWORD, VOCAB, HID, 0);
}
